// NeuralMemory_89687507076227
// MI455X (gfx1250) — matrix operand order repaired and checked by reading, not compiled, not run on hardware
//
#include <hip/hip_runtime.h>
#include <hip/hip_bf16.h>

// ---------------------------------------------------------------------------
// Types for CDNA5 WMMA (wave32, 16x16x32 bf16 -> f32)
// ---------------------------------------------------------------------------
typedef __bf16 bf16_t;
typedef bf16_t v16bf __attribute__((ext_vector_type(16)));
typedef float  v8f   __attribute__((ext_vector_type(8)));

union FragBF {
    v16bf v;
    uint4 q[2];
};

#define BB   4
#define TT   512
#define DIMM 512
#define NH   4
#define HD   64
#define HIDN 256
#define MROW (BB * TT)   // 2048

__device__ __forceinline__ unsigned short f2bf(float f) {
    union { float f; unsigned int u; } v;
    v.f = f;
    unsigned int u = v.u;
    unsigned int lsb = (u >> 16) & 1u;
    u += 0x7fffu + lsb;               // round-to-nearest-even
    return (unsigned short)(u >> 16);
}

// ---------------------------------------------------------------------------
// Elementwise f32 -> bf16 cast
// ---------------------------------------------------------------------------
__global__ void cast_bf16_kernel(const float* __restrict__ src,
                                 unsigned short* __restrict__ dst, int n) {
    int i = blockIdx.x * blockDim.x + threadIdx.x;
    if (i < n) dst[i] = f2bf(src[i]);
}

// ---------------------------------------------------------------------------
// C(MxN,f32) = A(MxK,bf16 row-major) @ B(NxK,bf16 row-major)^T
// Register-blocked: one 32x32 macro-tile per wave (2x2 of 16x16 WMMA tiles),
// so each K-step issues 8 b128 loads and 4 WMMAs (2x the reuse of 1-tile).
// ---------------------------------------------------------------------------
__global__ __launch_bounds__(256) void gemm_abt_bf16_kernel(
    const unsigned short* __restrict__ A,
    const unsigned short* __restrict__ Bm,
    float* __restrict__ C, int M, int N, int K) {
    const int lane = threadIdx.x & 31;
    const int wave = threadIdx.x >> 5;
    const int r    = lane & 15;
    const int half = lane >> 4;
    const int tilesN = N >> 5;                 // 32-wide macro tiles
    const int totTiles = (M >> 5) * tilesN;
    int tile = blockIdx.x * (blockDim.x >> 5) + wave;
    if (tile >= totTiles) return;
    const int tm = (tile / tilesN) << 5;
    const int tn = (tile % tilesN) << 5;

    v8f c00 = {0.f,0.f,0.f,0.f,0.f,0.f,0.f,0.f};
    v8f c01 = c00, c10 = c00, c11 = c00;

    const unsigned short* arow0 = A  + (size_t)(tm + r)      * K;
    const unsigned short* arow1 = A  + (size_t)(tm + 16 + r) * K;
    const unsigned short* brow0 = Bm + (size_t)(tn + r)      * K;
    const unsigned short* brow1 = Bm + (size_t)(tn + 16 + r) * K;

    for (int kk = 0; kk < K; kk += 32) {
        FragBF a0, a1, b0, b1;
        // A (16x32): lane r holds its row; lanes 0-15: K {0..7,16..23}, lanes 16-31: {8..15,24..31}
        a0.q[0] = *(const uint4*)(arow0 + kk + half * 8);
        a0.q[1] = *(const uint4*)(arow0 + kk + 16 + half * 8);
        a1.q[0] = *(const uint4*)(arow1 + kk + half * 8);
        a1.q[1] = *(const uint4*)(arow1 + kk + 16 + half * 8);
        // B (32x16): lane r holds its col; lanes 0-15: K {0..7,16..23}, lanes 16-31: {8..15,24..31}
        b0.q[0] = *(const uint4*)(brow0 + kk + half * 8);
        b0.q[1] = *(const uint4*)(brow0 + kk + 16 + half * 8);
        b1.q[0] = *(const uint4*)(brow1 + kk + half * 8);
        b1.q[1] = *(const uint4*)(brow1 + kk + 16 + half * 8);
        c00 = __builtin_amdgcn_wmma_f32_16x16x32_bf16(false, a0.v, false, b0.v, (short)0, c00, false, false);
        c01 = __builtin_amdgcn_wmma_f32_16x16x32_bf16(false, a0.v, false, b1.v, (short)0, c01, false, false);
        c10 = __builtin_amdgcn_wmma_f32_16x16x32_bf16(false, a1.v, false, b0.v, (short)0, c10, false, false);
        c11 = __builtin_amdgcn_wmma_f32_16x16x32_bf16(false, a1.v, false, b1.v, (short)0, c11, false, false);
    }
    // C/D layout: VGPR g -> row base + g + half*8, col base + r
    {
        float* p = C + (size_t)(tm + half * 8) * N + tn + r;
#pragma unroll
        for (int g = 0; g < 8; ++g) { p[(size_t)g * N] = c00[g]; p[(size_t)g * N + 16] = c01[g]; }
        p = C + (size_t)(tm + 16 + half * 8) * N + tn + r;
#pragma unroll
        for (int g = 0; g < 8; ++g) { p[(size_t)g * N] = c10[g]; p[(size_t)g * N + 16] = c11[g]; }
    }
}

// ---------------------------------------------------------------------------
// LayerNorm over head dim + transpose (B*T, H*d) -> (H, B, T, d)
// One 64-thread block per (b, t, h).
// ---------------------------------------------------------------------------
__global__ void ln_transpose_kernel(const float* __restrict__ P,
                                    const float* __restrict__ gamma,
                                    const float* __restrict__ beta,
                                    float* __restrict__ out) {
    __shared__ float red[64];
    const int i  = threadIdx.x;
    const int h  = blockIdx.x & (NH - 1);
    const int bt = blockIdx.x >> 2;
    const int b  = bt / TT;
    const int t  = bt - b * TT;

    float x = P[(size_t)bt * (NH * HD) + h * HD + i];
    red[i] = x; __syncthreads();
    for (int s = 32; s > 0; s >>= 1) { if (i < s) red[i] += red[i + s]; __syncthreads(); }
    float mu = red[0] * (1.0f / HD); __syncthreads();
    float d = x - mu;
    red[i] = d * d; __syncthreads();
    for (int s = 32; s > 0; s >>= 1) { if (i < s) red[i] += red[i + s]; __syncthreads(); }
    float var = red[0] * (1.0f / HD);
    float y = d * rsqrtf(var + 1e-5f) * gamma[i] + beta[i];
    out[(((size_t)(h * BB + b) * TT) + t) * HD + i] = y;
}

// ---------------------------------------------------------------------------
// Sequential fast-weight scan. One block per (h,b) chain; all fast-weight
// state (W1/M1/W2/M2 + biases) lives in LDS for the whole T=512 loop.
// Padded row strides (65 / 257 words) keep every access bank-conflict-free.
// Next-step k/v/q rows are prefetched (global_prefetch_b8) during compute.
// ---------------------------------------------------------------------------
#define S1 (HD + 1)      // 65
#define S2 (HIDN + 1)    // 257

__global__ __launch_bounds__(256, 1) void scan_kernel(
    const float* __restrict__ ks, const float* __restrict__ vs,
    const float* __restrict__ qs,
    const float* __restrict__ fc1_w, const float* __restrict__ fc1_b,
    const float* __restrict__ fc2_w, const float* __restrict__ fc2_b,
    const float* __restrict__ lr_scale, const float* __restrict__ mom_scale,
    float* __restrict__ ys) {
    extern __shared__ float sm[];
    float* W1  = sm;                    // HIDN * S1
    float* M1  = W1  + HIDN * S1;
    float* W2  = M1  + HIDN * S1;       // HD * S2   (row j = output dim)
    float* M2  = W2  + HD * S2;
    float* b1  = M2  + HD * S2;         // HIDN
    float* mb1 = b1  + HIDN;
    float* b2  = mb1 + HIDN;            // HD
    float* mb2 = b2  + HD;
    float* kt  = mb2 + HD;              // HD
    float* vt  = kt  + HD;
    float* qt  = vt  + HD;
    float* hv  = qt  + HD;              // HIDN
    float* av  = hv  + HIDN;
    float* dh  = av  + HIDN;
    float* dov = dh  + HIDN;            // HD
    float* aq  = dov + HD;              // HIDN

    const int tid = threadIdx.x;
    const int hb  = blockIdx.x;         // h*B + b
    const int h   = hb >> 2;
    const float lr  = 0.1f / (1.0f + __expf(-lr_scale[h]));
    const float mom = 0.9f / (1.0f + __expf(-mom_scale[h]));

    // Initialize fast weights from fc*_w[h]; momenta to zero.
    for (int idx = tid; idx < HIDN * HD; idx += blockDim.x) {
        int e = idx >> 6, i = idx & 63;
        W1[e * S1 + i] = fc1_w[(size_t)h * HIDN * HD + idx];
        M1[e * S1 + i] = 0.f;
    }
    for (int idx = tid; idx < HD * HIDN; idx += blockDim.x) {
        int j = idx >> 8, e = idx & 255;
        W2[j * S2 + e] = fc2_w[(size_t)h * HD * HIDN + idx];
        M2[j * S2 + e] = 0.f;
    }
    { int e = tid; b1[e] = fc1_b[h * HIDN + e]; mb1[e] = 0.f; }
    if (tid < HD) { b2[tid] = fc2_b[h * HD + tid]; mb2[tid] = 0.f; }
    __syncthreads();

    const float SQ2PI  = 0.79788456080286536f;
    const float INVSQ2 = 0.70710678118654752f;
    const size_t base  = (size_t)hb * TT * HD;

    for (int t = 0; t < TT; ++t) {
        if (tid < HD) {
            kt[tid] = ks[base + (size_t)t * HD + tid];
            vt[tid] = vs[base + (size_t)t * HD + tid];
            qt[tid] = qs[base + (size_t)t * HD + tid];
        }
        // prefetch next step's rows while this step computes
        if (t + 1 < TT && tid < HD) {
            __builtin_prefetch(&ks[base + (size_t)(t + 1) * HD + tid], 0, 3);
            __builtin_prefetch(&vs[base + (size_t)(t + 1) * HD + tid], 0, 3);
            __builtin_prefetch(&qs[base + (size_t)(t + 1) * HD + tid], 0, 3);
        }
        __syncthreads();
        // forward: h = W1 k + b1 ; a = gelu_exact(h)
        {
            const int e = tid;
            float s = b1[e];
#pragma unroll 8
            for (int i = 0; i < HD; ++i) s += W1[e * S1 + i] * kt[i];
            hv[e] = s;
            float cdf = 0.5f * (1.0f + erff(s * INVSQ2));
            av[e] = s * cdf;
        }
        __syncthreads();
        // out = W2 a + b2 ; d_out = out - v
        if (tid < HD) {
            float s = b2[tid];
#pragma unroll 8
            for (int e = 0; e < HIDN; ++e) s += W2[tid * S2 + e] * av[e];
            dov[tid] = s - vt[tid];
        }
        __syncthreads();
        // d_a = W2^T d_out ; d_h = d_a * (cdf + h*pdf)
        {
            const int e = tid;
            float s = 0.f;
#pragma unroll 8
            for (int j = 0; j < HD; ++j) s += W2[j * S2 + e] * dov[j];
            float hh  = hv[e];
            float cdf = 0.5f * (1.0f + erff(hh * INVSQ2));
            float pdf = SQ2PI * __expf(-0.5f * hh * hh);
            dh[e] = s * (cdf + hh * pdf);
        }
        __syncthreads();
        // momentum + SGD update (rank-1 grads)
        for (int idx = tid; idx < HIDN * HD; idx += blockDim.x) {
            int e = idx >> 6, i = idx & 63;
            float m = mom * M1[e * S1 + i] + dh[e] * kt[i];
            M1[e * S1 + i] = m;
            W1[e * S1 + i] -= lr * m;
        }
        for (int idx = tid; idx < HD * HIDN; idx += blockDim.x) {
            int j = idx >> 8, e = idx & 255;
            float m = mom * M2[j * S2 + e] + dov[j] * av[e];
            M2[j * S2 + e] = m;
            W2[j * S2 + e] -= lr * m;
        }
        { int e = tid; float m = mom * mb1[e] + dh[e]; mb1[e] = m; b1[e] -= lr * m; }
        if (tid < HD) { float m = mom * mb2[tid] + dov[tid]; mb2[tid] = m; b2[tid] -= lr * m; }
        __syncthreads();
        // retrieve with updated weights: y = mlp(P2, q)
        {
            const int e = tid;
            float s = b1[e];
#pragma unroll 8
            for (int i = 0; i < HD; ++i) s += W1[e * S1 + i] * qt[i];
            float cdf = 0.5f * (1.0f + erff(s * INVSQ2));
            aq[e] = s * cdf;
        }
        __syncthreads();
        if (tid < HD) {
            float s = b2[tid];
#pragma unroll 8
            for (int e = 0; e < HIDN; ++e) s += W2[tid * S2 + e] * aq[e];
            ys[base + (size_t)t * HD + tid] = s;
        }
        __syncthreads();
    }
}

// ---------------------------------------------------------------------------
// ys (H,B,T,d) f32 -> (B*T, H*d) bf16 for the output GEMM
// ---------------------------------------------------------------------------
__global__ void ycast_kernel(const float* __restrict__ ys,
                             unsigned short* __restrict__ ybf) {
    int idx = blockIdx.x * blockDim.x + threadIdx.x;   // H*B*T*d = 524288
    int i    = idx & (HD - 1);
    int rest = idx >> 6;                 // hb*T + t
    int t    = rest & (TT - 1);
    int hb   = rest >> 9;
    int b    = hb & (BB - 1);
    int h    = hb >> 2;
    ybf[((size_t)(b * TT + t)) * (NH * HD) + h * HD + i] = f2bf(ys[idx]);
}

// ---------------------------------------------------------------------------
// out = sigmoid(glog + gate_b) * r
// ---------------------------------------------------------------------------
__global__ void epilogue_kernel(const float* __restrict__ glog,
                                const float* __restrict__ rbuf,
                                const float* __restrict__ gate_b,
                                float* __restrict__ out) {
    int idx = blockIdx.x * blockDim.x + threadIdx.x;   // M*512
    int n = idx & (DIMM - 1);
    float g = 1.0f / (1.0f + __expf(-(glog[idx] + gate_b[n])));
    out[idx] = g * rbuf[idx];
}

// ---------------------------------------------------------------------------
// Host-side orchestration
// ---------------------------------------------------------------------------
extern "C" void kernel_launch(void* const* d_in, const int* in_sizes, int n_in,
                              void* d_out, int out_size, void* d_ws, size_t ws_size,
                              hipStream_t stream) {
    (void)in_sizes; (void)n_in; (void)out_size; (void)ws_size;
    const float* x       = (const float*)d_in[0];
    const float* w_k     = (const float*)d_in[1];
    const float* w_v     = (const float*)d_in[2];
    const float* w_q     = (const float*)d_in[3];
    const float* w_o     = (const float*)d_in[4];
    const float* ln_k_g  = (const float*)d_in[5];
    const float* ln_k_b  = (const float*)d_in[6];
    const float* ln_v_g  = (const float*)d_in[7];
    const float* ln_v_b  = (const float*)d_in[8];
    const float* ln_q_g  = (const float*)d_in[9];
    const float* ln_q_b  = (const float*)d_in[10];
    const float* fc1_w   = (const float*)d_in[11];
    const float* fc1_b   = (const float*)d_in[12];
    const float* fc2_w   = (const float*)d_in[13];
    const float* fc2_b   = (const float*)d_in[14];
    const float* lr_s    = (const float*)d_in[15];
    const float* mom_s   = (const float*)d_in[16];
    const float* gate_w  = (const float*)d_in[17];
    const float* gate_b  = (const float*)d_in[18];
    float* out = (float*)d_out;

    char* wp = (char*)d_ws;
    auto alloc = [&](size_t bytes) -> char* {
        char* p = wp;
        wp += (bytes + 255) & ~(size_t)255;
        return p;
    };
    unsigned short* xb  = (unsigned short*)alloc((size_t)MROW * DIMM * 2);
    unsigned short* wkb = (unsigned short*)alloc((size_t)NH * HD * DIMM * 2);
    unsigned short* wvb = (unsigned short*)alloc((size_t)NH * HD * DIMM * 2);
    unsigned short* wqb = (unsigned short*)alloc((size_t)NH * HD * DIMM * 2);
    unsigned short* wob = (unsigned short*)alloc((size_t)DIMM * NH * HD * 2);
    unsigned short* gwb = (unsigned short*)alloc((size_t)DIMM * DIMM * 2);
    float* proj = (float*)alloc((size_t)MROW * NH * HD * 4);
    float* ksb  = (float*)alloc((size_t)NH * BB * TT * HD * 4);
    float* vsb  = (float*)alloc((size_t)NH * BB * TT * HD * 4);
    float* qsb  = (float*)alloc((size_t)NH * BB * TT * HD * 4);
    float* ysb  = (float*)alloc((size_t)NH * BB * TT * HD * 4);
    unsigned short* ybf = (unsigned short*)alloc((size_t)MROW * NH * HD * 2);
    float* rbuf = (float*)alloc((size_t)MROW * DIMM * 4);
    float* glog = (float*)alloc((size_t)MROW * DIMM * 4);

    // --- casts to bf16 ---
    int nx = MROW * DIMM;
    cast_bf16_kernel<<<(nx + 255) / 256, 256, 0, stream>>>(x, xb, nx);
    int nw = NH * HD * DIMM;
    cast_bf16_kernel<<<(nw + 255) / 256, 256, 0, stream>>>(w_k, wkb, nw);
    cast_bf16_kernel<<<(nw + 255) / 256, 256, 0, stream>>>(w_v, wvb, nw);
    cast_bf16_kernel<<<(nw + 255) / 256, 256, 0, stream>>>(w_q, wqb, nw);
    int no = DIMM * NH * HD;
    cast_bf16_kernel<<<(no + 255) / 256, 256, 0, stream>>>(w_o, wob, no);
    int ng = DIMM * DIMM;
    cast_bf16_kernel<<<(ng + 255) / 256, 256, 0, stream>>>(gate_w, gwb, ng);

    // --- projections + LN + transpose ---
    const int N1 = NH * HD;                          // 256
    int tiles1 = (MROW / 32) * (N1 / 32);            // 512 macro-tiles
    gemm_abt_bf16_kernel<<<tiles1 / 8, 256, 0, stream>>>(xb, wkb, proj, MROW, N1, DIMM);
    ln_transpose_kernel<<<MROW * NH, 64, 0, stream>>>(proj, ln_k_g, ln_k_b, ksb);
    gemm_abt_bf16_kernel<<<tiles1 / 8, 256, 0, stream>>>(xb, wvb, proj, MROW, N1, DIMM);
    ln_transpose_kernel<<<MROW * NH, 64, 0, stream>>>(proj, ln_v_g, ln_v_b, vsb);
    gemm_abt_bf16_kernel<<<tiles1 / 8, 256, 0, stream>>>(xb, wqb, proj, MROW, N1, DIMM);
    ln_transpose_kernel<<<MROW * NH, 64, 0, stream>>>(proj, ln_q_g, ln_q_b, qsb);

    // --- gate logits GEMM ---
    int tilesg = (MROW / 32) * (DIMM / 32);          // 1024 macro-tiles
    gemm_abt_bf16_kernel<<<tilesg / 8, 256, 0, stream>>>(xb, gwb, glog, MROW, DIMM, DIMM);

    // --- sequential fast-weight scan: 16 chains, LDS-resident state ---
    size_t smem = (size_t)(2 * (HIDN * S1) + 2 * (HD * S2)
                           + 2 * HIDN + 2 * HD      // b1/mb1, b2/mb2
                           + 3 * HD                 // kt, vt, qt
                           + 3 * HIDN + HD + HIDN   // hv, av, dh, dov, aq
                          ) * sizeof(float);
    scan_kernel<<<NH * BB, 256, smem, stream>>>(ksb, vsb, qsb, fc1_w, fc1_b,
                                                fc2_w, fc2_b, lr_s, mom_s, ysb);

    // --- output projection + gated epilogue ---
    int ny = NH * BB * TT * HD;
    ycast_kernel<<<(ny + 255) / 256, 256, 0, stream>>>(ysb, ybf);
    int tileso = (MROW / 32) * (DIMM / 32);
    gemm_abt_bf16_kernel<<<tileso / 8, 256, 0, stream>>>(ybf, wob, rbuf, MROW, DIMM, N1);
    int ne = MROW * DIMM;
    epilogue_kernel<<<(ne + 255) / 256, 256, 0, stream>>>(glog, rbuf, gate_b, out);
}
